// GumbelSampler_42674795053920
// MI455X (gfx1250) — matrix operand order repaired and checked by reading, not compiled, not run on hardware
//
#include <hip/hip_runtime.h>
#include <stdint.h>
#include <string.h>

// ---------------------------------------------------------------------------
// GumbelSampler for MI455X (gfx1250, wave32, WMMA)
//   reps:[32,2048,768] f32, mask:[32,2048] bool, gumbel:[32,2048] f32
//   W1:[384,768], b1:[384], W2:[1,384], b2:[1]
// Outputs (concat, as float): sampled_reps [32,2048,768],
//   sampled_mask [32,2048], ind [32,128]
// ---------------------------------------------------------------------------

typedef __attribute__((ext_vector_type(16))) _Float16 v16h;
typedef __attribute__((ext_vector_type(8)))  float    v8f;

#define B_      32
#define N_      2048
#define D_      768
#define H_      384
#define TOPK_   128
#define ROWS_   (B_ * N_)        // 65536
#define KT_     32               // WMMA K per step (f16)
#define HT_     16               // WMMA N per tile
#define KFRAGS_ (D_ / KT_)       // 24
#define HFRAGS_ (H_ / HT_)       // 24
#define INV_TEMP 10.0f

union H2 { _Float16 h[2]; uint32_t u; };
union BFrag { uint4 u[2]; v16h h; };

// Hardware tanh: CDNA5 V_TANH_F32 is a 1-cycle-class TRANS32 op vs ~25-op
// ocml polynomial. TRANS hazard rule: 1 independent op before result use ->
// compiler handles for the builtin; inline-asm path adds an explicit v_nop.
#if __has_builtin(__builtin_amdgcn_tanhf)
__device__ __forceinline__ float fast_tanh(float x) {
    return __builtin_amdgcn_tanhf(x);
}
#else
__device__ __forceinline__ float fast_tanh(float x) {
    float r;
    asm volatile("v_tanh_f32 %0, %1\n\tv_nop" : "=v"(r) : "v"(x));
    return r;
}
#endif

// ---------------------------------------------------------------------------
// K0: repack W1 (f32 [H,D] row-major) into f16 B-operand fragments.
// Fragment (ht,kt): 32 lanes x 8 dwords, lane-contiguous 32B blocks so the
// GEMM does two coalesced b128 loads per lane. Layout: lane<16 holds column
// n=lane, K=kt*32..+15; lane>=16 holds same column set, K=+16..+31 (contiguous k order,
// same order as the A fragment in K1).
// ---------------------------------------------------------------------------
__global__ __launch_bounds__(32)
void pack_w1(const float* __restrict__ W1, uint32_t* __restrict__ wpack) {
    const int frag = blockIdx.x;             // ht*KFRAGS_ + kt
    const int ht = frag / KFRAGS_;
    const int kt = frag % KFRAGS_;
    const int l  = threadIdx.x;              // 0..31
    const int n  = ht * HT_ + (l & 15);
    const int kb = kt * KT_ + ((l >> 4) << 4);   // 0 or +16
    uint32_t out[8];
#pragma unroll
    for (int j = 0; j < 8; ++j) {
        H2 t;
        t.h[0] = (_Float16)W1[(size_t)n * D_ + kb + 2 * j];
        t.h[1] = (_Float16)W1[(size_t)n * D_ + kb + 2 * j + 1];
        out[j] = t.u;
    }
    uint4* dst = (uint4*)(wpack + ((size_t)frag * 32 + l) * 8);
    dst[0] = make_uint4(out[0], out[1], out[2], out[3]);
    dst[1] = make_uint4(out[4], out[5], out[6], out[7]);
}

// ---------------------------------------------------------------------------
// K1: fused logits = W2 . tanh(W1 . reps[row] + b1) + b2
// One wave per 16 rows. kt-outer / ht-inner: 24 register-resident v8f
// accumulators, reps streamed from HBM exactly once (f32 -> f16 in-register),
// W1 fragments stream from L2 (0.59 MB resident).
// ---------------------------------------------------------------------------
__global__ __launch_bounds__(256)
void logits_kernel(const float* __restrict__ reps,
                   const uint32_t* __restrict__ wpack,
                   const float* __restrict__ b1,
                   const float* __restrict__ W2,
                   const float* __restrict__ b2,
                   float* __restrict__ logits) {
    const int wave    = threadIdx.x >> 5;          // 0..7
    const int lane    = threadIdx.x & 31;
    const int rowBase = blockIdx.x * 128 + wave * 16;
    const int halfSel = lane >> 4;                 // 0 | 1
    const float* arow = reps + (size_t)(rowBase + (lane & 15)) * D_;

    v8f acc[HFRAGS_];
#pragma unroll
    for (int t = 0; t < HFRAGS_; ++t) acc[t] = (v8f){};

    for (int kt = 0; kt < KFRAGS_; ++kt) {
        // ---- A fragment: 16 f32 from this lane's row, contiguous k order as packed W1
        // lanes<16 take K {0..15}, lanes>=16 take K {16..31}.
        const int c0 = kt * KT_ + halfSel * 16;
        const int c1 = c0 + 8;
        if (kt + 1 < KFRAGS_)
            __builtin_prefetch(arow + c0 + KT_, 0, 0);  // global_prefetch_b8
        float4 f0 = *(const float4*)(arow + c0);
        float4 f1 = *(const float4*)(arow + c0 + 4);
        float4 f2 = *(const float4*)(arow + c1);
        float4 f3 = *(const float4*)(arow + c1 + 4);
        v16h a;
        a[0]  = (_Float16)f0.x; a[1]  = (_Float16)f0.y;
        a[2]  = (_Float16)f0.z; a[3]  = (_Float16)f0.w;
        a[4]  = (_Float16)f1.x; a[5]  = (_Float16)f1.y;
        a[6]  = (_Float16)f1.z; a[7]  = (_Float16)f1.w;
        a[8]  = (_Float16)f2.x; a[9]  = (_Float16)f2.y;
        a[10] = (_Float16)f2.z; a[11] = (_Float16)f2.w;
        a[12] = (_Float16)f3.x; a[13] = (_Float16)f3.y;
        a[14] = (_Float16)f3.z; a[15] = (_Float16)f3.w;

#pragma unroll
        for (int ht = 0; ht < HFRAGS_; ++ht) {
            const uint32_t* bp =
                wpack + (((size_t)(ht * KFRAGS_ + kt)) * 32 + lane) * 8;
            BFrag bb;
            bb.u[0] = *(const uint4*)(bp);
            bb.u[1] = *(const uint4*)(bp + 4);
            acc[ht] = __builtin_amdgcn_wmma_f32_16x16x32_f16(
                false, a, false, bb.h, (short)0, acc[ht], false, false);
        }
    }

    // ---- epilogue: tanh + b1, dot with W2, per-row reduce
    // C/D layout: lane%16 = column, VGPR r -> row r (lanes<16) / r+8 (>=16).
    float lsum[8];
#pragma unroll
    for (int r = 0; r < 8; ++r) lsum[r] = 0.0f;
#pragma unroll
    for (int ht = 0; ht < HFRAGS_; ++ht) {
        const int hcol = ht * HT_ + (lane & 15);
        const float w2v = W2[hcol];
        const float b1v = b1[hcol];
#pragma unroll
        for (int r = 0; r < 8; ++r)
            lsum[r] += fast_tanh(acc[ht][r] + b1v) * w2v;
    }
    const float bias = b2[0];
#pragma unroll
    for (int r = 0; r < 8; ++r) {
        float v = lsum[r];
#pragma unroll
        for (int off = 1; off < 16; off <<= 1)
            v += __shfl_xor(v, off, 16);       // reduce the 16 columns
        if ((lane & 15) == 0)
            logits[rowBase + r + 8 * halfSel] = v + bias;
    }
}

// ---------------------------------------------------------------------------
// K2: per-batch gumbel-softmax + top-k (bitonic sort of 2048 in LDS).
// Emits ind (as float), sampled_mask (0/1 float), and gumbel_mask row scale.
// ---------------------------------------------------------------------------
__global__ __launch_bounds__(256)
void softmax_topk(const float* __restrict__ logits,
                  const uint8_t* __restrict__ mask,
                  const float* __restrict__ gnoise,
                  float* __restrict__ out_mask,
                  float* __restrict__ out_ind,
                  float* __restrict__ ws_gm) {
    const int b   = blockIdx.x;
    const int tid = threadIdx.x;
    __shared__ float s_y[N_];
    __shared__ float s_val[N_];
    __shared__ int   s_idx[N_];
    __shared__ float s_red[256];

    // scaled logits
    float lmax = -3.4e38f;
    for (int n = tid; n < N_; n += 256) {
        float x = logits[b * N_ + n] + gnoise[b * N_ + n] +
                  (mask[b * N_ + n] ? 0.0f : -10000.0f);
        x *= INV_TEMP;
        s_y[n] = x;
        lmax = fmaxf(lmax, x);
    }
    s_red[tid] = lmax;
    __syncthreads();
    for (int s = 128; s > 0; s >>= 1) {
        if (tid < s) s_red[tid] = fmaxf(s_red[tid], s_red[tid + s]);
        __syncthreads();
    }
    const float bmax = s_red[0];
    __syncthreads();

    float lsumv = 0.0f;
    for (int n = tid; n < N_; n += 256) {
        float e = __expf(s_y[n] - bmax);
        s_y[n] = e;
        lsumv += e;
    }
    s_red[tid] = lsumv;
    __syncthreads();
    for (int s = 128; s > 0; s >>= 1) {
        if (tid < s) s_red[tid] += s_red[tid + s];
        __syncthreads();
    }
    const float inv = 1.0f / s_red[0];
    __syncthreads();
    for (int n = tid; n < N_; n += 256) s_y[n] *= inv;
    __syncthreads();
    if (tid == 0) s_y[0] += 10000.0f;     // force class 0 into top-k
    __syncthreads();

    for (int n = tid; n < N_; n += 256) { s_val[n] = s_y[n]; s_idx[n] = n; }
    __syncthreads();

    // bitonic sort, descending
    for (int k = 2; k <= N_; k <<= 1) {
        for (int j = k >> 1; j > 0; j >>= 1) {
            for (int t = tid; t < N_; t += 256) {
                const int ixj = t ^ j;
                if (ixj > t) {
                    const bool desc = ((t & k) == 0);
                    float v0 = s_val[t], v1 = s_val[ixj];
                    if (desc ? (v0 < v1) : (v0 > v1)) {
                        s_val[t] = v1; s_val[ixj] = v0;
                        int ti = s_idx[t]; s_idx[t] = s_idx[ixj]; s_idx[ixj] = ti;
                    }
                }
            }
            __syncthreads();
        }
    }

    int myind = -1;
    if (tid < TOPK_) myind = s_idx[tid];
    __syncthreads();
    for (int n = tid; n < N_; n += 256) s_val[n] = 0.0f;  // reuse as gm
    __syncthreads();
    if (tid < TOPK_) {
        const float y = s_y[myind];
        s_val[myind] = (1.0f - y) + y;     // straight-through forward value
        out_ind[b * TOPK_ + tid] = (float)myind;
    }
    __syncthreads();
    for (int n = tid; n < N_; n += 256) {
        const float gm = s_val[n];
        ws_gm[b * N_ + n] = gm;
        out_mask[b * N_ + n] = (mask[b * N_ + n] && gm != 0.0f) ? 1.0f : 0.0f;
    }
}

// ---------------------------------------------------------------------------
// K3: sampled_reps = reps * gm[row]; skip the HBM read for zeroed rows
// (only 128/2048 rows survive -> 12.6 MB read + 201 MB write).
// ---------------------------------------------------------------------------
__global__ __launch_bounds__(256)
void scale_reps(const float* __restrict__ reps,
                const float* __restrict__ ws_gm,
                float* __restrict__ out) {
    const size_t f = (size_t)blockIdx.x * 256 + threadIdx.x;  // float4 index
    const size_t e = f * 4;
    const int row = (int)(e / D_);
    const float gm = ws_gm[row];
    float4 o;
    if (gm == 0.0f) {
        o = make_float4(0.0f, 0.0f, 0.0f, 0.0f);
    } else {
        float4 r = *(const float4*)(reps + e);
        o = make_float4(r.x * gm, r.y * gm, r.z * gm, r.w * gm);
    }
    *(float4*)(out + e) = o;
}

// ---------------------------------------------------------------------------
extern "C" void kernel_launch(void* const* d_in, const int* in_sizes, int n_in,
                              void* d_out, int out_size, void* d_ws,
                              size_t ws_size, hipStream_t stream) {
    const float*   reps   = (const float*)d_in[0];
    const uint8_t* mask   = (const uint8_t*)d_in[1];
    const float*   gnoise = (const float*)d_in[2];
    const float*   W1     = (const float*)d_in[3];
    const float*   b1     = (const float*)d_in[4];
    const float*   W2     = (const float*)d_in[5];
    const float*   b2     = (const float*)d_in[6];

    float* out      = (float*)d_out;
    float* out_reps = out;                                   // B*N*D
    float* out_mask = out + (size_t)ROWS_ * D_;              // B*N
    float* out_ind  = out_mask + ROWS_;                      // B*TOPK

    // workspace: packed W1 fragments | logits | gumbel_mask
    const size_t WPACK_DW = (size_t)HFRAGS_ * KFRAGS_ * 32 * 8;  // 147456 dw
    uint32_t* wpack     = (uint32_t*)d_ws;
    float*    ws_logits = (float*)((char*)d_ws + WPACK_DW * 4);
    float*    ws_gm     = ws_logits + ROWS_;

    pack_w1<<<HFRAGS_ * KFRAGS_, 32, 0, stream>>>(W1, wpack);
    logits_kernel<<<ROWS_ / 128, 256, 0, stream>>>(reps, wpack, b1, W2, b2,
                                                   ws_logits);
    softmax_topk<<<B_, 256, 0, stream>>>(ws_logits, mask, gnoise, out_mask,
                                         out_ind, ws_gm);
    scale_reps<<<(ROWS_ * (D_ / 4)) / 256, 256, 0, stream>>>(reps, ws_gm,
                                                             out_reps);
}
